// OLSTM_82162724372548
// MI455X (gfx1250) — compile-verified, matrix operand order checked by reading, not run on hardware
//
#include <hip/hip_runtime.h>
#include <hip/hip_bf16.h>
#include <math.h>

typedef __attribute__((ext_vector_type(16))) _Float16 v16h;
typedef __attribute__((ext_vector_type(8)))  float    v8f;

#define EMBD    64
#define HID     128
#define INDIM   100
#define KPAD    128
#define NGATE   512
#define OBSLEN  2048
#define NNEIGH  4096
#define STEPS1  2047
#define STEPS2  2047
#define MROWS   2048
#define GRIDN   6
#define NB_BYTES (NNEIGH * 2 * 4)     /* 32 KB of neighbors per step */

__device__ __forceinline__ float sigmoidf_(float v) { return 1.0f / (1.0f + __expf(-v)); }

// ---------------------------------------------------------------------------
// CDNA5 async global->LDS copy helpers (ASYNCcnt-tracked, ISA 10.7/15.18.3).
// Each call: 8 x global_load_async_to_lds_b128 per wave => 32KB per 256-thread
// workgroup. LDS byte offset = low 32 bits of the generic pointer (ISA 10.2).
// ---------------------------------------------------------------------------
__device__ __forceinline__ void issue_nb_copy(const char* gsrc, char* ldst, int tid) {
    #pragma unroll
    for (int i = 0; i < 8; ++i) {
        const void* g = gsrc + (size_t)i * 4096 + (size_t)tid * 16;
        unsigned int l = (unsigned int)(uintptr_t)(ldst + (size_t)i * 4096 + (size_t)tid * 16);
        asm volatile("global_load_async_to_lds_b128 %0, %1, off"
                     :: "v"(l), "v"(g) : "memory");
    }
}
__device__ __forceinline__ void wait_async_le8() { asm volatile("s_wait_asynccnt 0x8" ::: "memory"); }
__device__ __forceinline__ void wait_async_0()  { asm volatile("s_wait_asynccnt 0x0" ::: "memory"); }

// ---------------------------------------------------------------------------
// Kernel 1: per-timestep phase-1 input row  Xh[t] = [relu(W_emb@(o2-o1)+b), occ(center=o2), 0-pad]
// ---------------------------------------------------------------------------
__global__ void k_build_x(const float* __restrict__ observed,
                          const float* __restrict__ other_paths,
                          const float* __restrict__ W_emb,
                          const float* __restrict__ b_emb,
                          _Float16* __restrict__ Xh) {
    const int t   = blockIdx.x;
    const int tid = threadIdx.x;
    const size_t ro = (size_t)t * KPAD;
    if (t >= STEPS1) {                       // zero pad row for the 2048-row GEMM
        if (tid < KPAD) Xh[ro + tid] = (_Float16)0.0f;
        return;
    }
    __shared__ float occ[GRIDN * GRIDN];
    if (tid < GRIDN * GRIDN) occ[tid] = 0.0f;
    const float cx = observed[2 * (t + 1) + 0];
    const float cy = observed[2 * (t + 1) + 1];
    __syncthreads();

    const float* nb = other_paths + (size_t)(t + 1) * (size_t)NNEIGH * 2;
    #pragma unroll 4
    for (int i = 0; i < NNEIGH / 256; ++i) {
        const int idx = tid * (NNEIGH / 256) + i;
        const float rx = nb[2 * idx + 0] - cx;
        const float ry = nb[2 * idx + 1] - cy;
        const int ix = (int)floorf((rx + 1.5f) * 2.0f);
        const int iy = (int)floorf((ry + 1.5f) * 2.0f);
        if (ix >= 0 && ix < GRIDN && iy >= 0 && iy < GRIDN) {
            const float gx0 = -1.5f + ix * 0.5f;
            const float gy0 = -1.5f + iy * 0.5f;
            if (rx > gx0 && rx < gx0 + 0.5f && ry > gy0 && ry < gy0 + 0.5f)
                occ[ix * GRIDN + iy] = 1.0f;     // benign race: all writers store 1.0
        }
    }
    __syncthreads();

    const float dx = observed[2 * (t + 1) + 0] - observed[2 * t + 0];
    const float dy = observed[2 * (t + 1) + 1] - observed[2 * t + 1];
    if (tid < EMBD) {
        const float e = W_emb[2 * tid] * dx + W_emb[2 * tid + 1] * dy + b_emb[tid];
        Xh[ro + tid] = (_Float16)fmaxf(e, 0.0f);
    } else if (tid < EMBD + GRIDN * GRIDN) {
        Xh[ro + tid] = (_Float16)occ[tid - EMBD];
    } else if (tid < KPAD) {
        Xh[ro + tid] = (_Float16)0.0f;
    }
}

// ---------------------------------------------------------------------------
// Kernel 2: W_ih (512x100 f32) -> Wh (512x128 f16, zero-padded K)
// ---------------------------------------------------------------------------
__global__ void k_conv_w(const float* __restrict__ W_ih, _Float16* __restrict__ Wh) {
    const int idx = blockIdx.x * blockDim.x + threadIdx.x;   // 0..65535
    const int j = idx >> 7, k = idx & 127;
    Wh[idx] = (k < INDIM) ? (_Float16)W_ih[j * INDIM + k] : (_Float16)0.0f;
}

// ---------------------------------------------------------------------------
// Kernel 3: WMMA GEMM  G1(2048x512) = Xh(2048x128) @ Whᵀ + b_ih
// One wave per 16x16 output tile; K-loop of 4 x v_wmma_f32_16x16x32_f16.
// ---------------------------------------------------------------------------
__global__ void k_gates_gemm(const _Float16* __restrict__ Xh,
                             const _Float16* __restrict__ Wh,
                             const float* __restrict__ b_ih,
                             float* __restrict__ G1) {
    const int lane  = threadIdx.x & 31;
    const int wave  = (blockIdx.x * blockDim.x + threadIdx.x) >> 5;
    const int mTile = wave >> 5;        // 0..127  (2048/16)
    const int nTile = wave & 31;        // 0..31   (512/16)
    const int r16   = lane & 15;
    const int hi    = (lane >> 4) & 1;

    const _Float16* arow = Xh + (size_t)(mTile * 16 + r16) * KPAD;
    const _Float16* brow = Wh + (size_t)(nTile * 16 + r16) * KPAD;

    v8f acc = {};
    #pragma unroll
    for (int kk = 0; kk < 4; ++kk) {
        const int kOff = kk * 32;
        v16h a, b;
        #pragma unroll
        for (int v = 0; v < 8; ++v) {
            const int k0 = kOff + (v & 3) * 2 + hi * 8 + (v >> 2) * 16;
            a[2 * v]     = arow[k0];
            a[2 * v + 1] = arow[k0 + 1];
            b[2 * v]     = brow[k0];
            b[2 * v + 1] = brow[k0 + 1];
        }
        acc = __builtin_amdgcn_wmma_f32_16x16x32_f16(
            false, a, false, b, (short)0, acc, false, false);
    }

    const int col  = nTile * 16 + r16;
    const float bi = b_ih[col];
    #pragma unroll
    for (int r = 0; r < 8; ++r) {
        const int row = mTile * 16 + hi * 8 + r;      // pad row 2047 stored, never read
        G1[(size_t)row * NGATE + col] = acc[r] + bi;
    }
}

// ---------------------------------------------------------------------------
// Kernel 4: serial 4094-step LSTM scan, single workgroup of 256 threads.
// Phase 1 consumes precomputed G1 (with next-row prefetch); phase 2 streams
// neighbors via double-buffered async global->LDS copies (ASYNCcnt), keeping
// HBM latency off the serial dependency chain.
// ---------------------------------------------------------------------------
__global__ void __launch_bounds__(256, 1)
k_scan(const float* __restrict__ observed,
       const float* __restrict__ other_paths,
       const float* __restrict__ W_emb, const float* __restrict__ b_emb,
       const float* __restrict__ W_ih,  const float* __restrict__ b_ih,
       const float* __restrict__ W_hh,  const float* __restrict__ b_hh,
       const float* __restrict__ W_out, const float* __restrict__ b_out,
       const float* __restrict__ G1,
       float* __restrict__ out) {
    extern __shared__ char nbuf[];        // 2 x 32KB neighbor double buffer (dynamic LDS)
    __shared__ float h[HID], c[HID], g[NGATE], xv[EMBD + GRIDN * GRIDN];
    __shared__ float s_n0, s_n1;
    __shared__ float pp[2], pc[2];
    const int tid = threadIdx.x;
    if (tid < HID) { h[tid] = 0.0f; c[tid] = 0.0f; }

    // Kick off the phase-2 t=0 neighbor copy now; it has all of phase 1 to land.
    const char* nb_g0 = (const char*)(other_paths + (size_t)OBSLEN * (size_t)NNEIGH * 2);
    issue_nb_copy(nb_g0, nbuf, tid);
    __syncthreads();

    // ---------------- phase 1 ----------------
    for (int t = 0; t < STEPS1; ++t) {
        const float* Gt = G1 + (size_t)t * NGATE;
        if (t + 1 < STEPS1)
            __builtin_prefetch(Gt + NGATE + tid * 2, 0, 1);
        #pragma unroll
        for (int jj = 0; jj < 2; ++jj) {
            const int j = tid + jj * 256;
            float acc = Gt[j] + b_hh[j];
            const float* wr = W_hh + (size_t)j * HID;
            #pragma unroll 8
            for (int k = 0; k < HID; ++k) acc += wr[k] * h[k];
            g[j] = acc;
        }
        __syncthreads();
        if (tid < HID) {
            const float iv = sigmoidf_(g[tid]);
            const float fv = sigmoidf_(g[HID + tid]);
            const float gv = tanhf(g[2 * HID + tid]);
            const float ov = sigmoidf_(g[3 * HID + tid]);
            const float cn = fv * c[tid] + iv * gv;
            c[tid] = cn;
            h[tid] = ov * tanhf(cn);
        }
        __syncthreads();
        if (tid < 5) {
            float acc = b_out[tid];
            const float* wr = W_out + tid * HID;
            #pragma unroll 8
            for (int k = 0; k < HID; ++k) acc += wr[k] * h[k];
            out[(size_t)t * 5 + tid] = acc;
            if (tid == 0) s_n0 = acc;
            if (tid == 1) s_n1 = acc;
        }
        __syncthreads();
        if (tid == 0) {
            pp[0] = pc[0]; pp[1] = pc[1];                        // pos history shift
            pc[0] = observed[2 * (t + 1) + 0] + s_n0;            // positions1[t]
            pc[1] = observed[2 * (t + 1) + 1] + s_n1;
        }
        __syncthreads();
    }
    // here pp = positions1[-2], pc = positions1[-1]

    // ---------------- phase 2 ----------------
    for (int t = 0; t < STEPS2; ++t) {
        // issue next step's async copy, then wait for the current buffer
        if (t + 1 < STEPS2) {
            issue_nb_copy(nb_g0 + (size_t)(t + 1) * NB_BYTES,
                          nbuf + (size_t)((t + 1) & 1) * NB_BYTES, tid);
            wait_async_le8();
        } else {
            wait_async_0();
        }
        const float cx = pc[0], cy = pc[1];
        if (tid < EMBD) {
            const float dx = pc[0] - pp[0];
            const float dy = pc[1] - pp[1];
            const float e  = W_emb[2 * tid] * dx + W_emb[2 * tid + 1] * dy + b_emb[tid];
            xv[tid] = fmaxf(e, 0.0f);
        } else if (tid < EMBD + GRIDN * GRIDN) {
            xv[tid] = 0.0f;
        }
        __syncthreads();   // copies visible + xv cleared
        const float2* nbl = (const float2*)(nbuf + (size_t)(t & 1) * NB_BYTES);
        #pragma unroll 4
        for (int i = 0; i < NNEIGH / 256; ++i) {
            const float2 p = nbl[i * 256 + tid];   // consecutive lanes -> consecutive 8B: conflict-free
            const float rx = p.x - cx;
            const float ry = p.y - cy;
            const int ix = (int)floorf((rx + 1.5f) * 2.0f);
            const int iy = (int)floorf((ry + 1.5f) * 2.0f);
            if (ix >= 0 && ix < GRIDN && iy >= 0 && iy < GRIDN) {
                const float gx0 = -1.5f + ix * 0.5f;
                const float gy0 = -1.5f + iy * 0.5f;
                if (rx > gx0 && rx < gx0 + 0.5f && ry > gy0 && ry < gy0 + 0.5f)
                    xv[EMBD + ix * GRIDN + iy] = 1.0f;
            }
        }
        __syncthreads();
        #pragma unroll
        for (int jj = 0; jj < 2; ++jj) {
            const int j = tid + jj * 256;
            float acc = b_ih[j] + b_hh[j];
            const float* wi = W_ih + (size_t)j * INDIM;
            #pragma unroll 4
            for (int k = 0; k < INDIM; ++k) acc += wi[k] * xv[k];
            const float* wr = W_hh + (size_t)j * HID;
            #pragma unroll 8
            for (int k = 0; k < HID; ++k) acc += wr[k] * h[k];
            g[j] = acc;
        }
        __syncthreads();
        if (tid < HID) {
            const float iv = sigmoidf_(g[tid]);
            const float fv = sigmoidf_(g[HID + tid]);
            const float gv = tanhf(g[2 * HID + tid]);
            const float ov = sigmoidf_(g[3 * HID + tid]);
            const float cn = fv * c[tid] + iv * gv;
            c[tid] = cn;
            h[tid] = ov * tanhf(cn);
        }
        __syncthreads();
        if (tid < 5) {
            float acc = b_out[tid];
            const float* wr = W_out + tid * HID;
            #pragma unroll 8
            for (int k = 0; k < HID; ++k) acc += wr[k] * h[k];
            out[(size_t)(STEPS1 + t) * 5 + tid] = acc;
            if (tid == 0) s_n0 = acc;
            if (tid == 1) s_n1 = acc;
        }
        __syncthreads();
        if (tid == 0) {
            const float nx = pc[0] + s_n0;
            const float ny = pc[1] + s_n1;
            pp[0] = pc[0]; pp[1] = pc[1];
            pc[0] = nx;    pc[1] = ny;
        }
        __syncthreads();
    }
}

// ---------------------------------------------------------------------------
extern "C" void kernel_launch(void* const* d_in, const int* in_sizes, int n_in,
                              void* d_out, int out_size, void* d_ws, size_t ws_size,
                              hipStream_t stream) {
    (void)in_sizes; (void)n_in; (void)out_size; (void)ws_size;
    const float* observed    = (const float*)d_in[0];
    const float* other_paths = (const float*)d_in[1];
    const float* W_emb = (const float*)d_in[2];
    const float* b_emb = (const float*)d_in[3];
    const float* W_ih  = (const float*)d_in[4];
    const float* b_ih  = (const float*)d_in[5];
    const float* W_hh  = (const float*)d_in[6];
    const float* b_hh  = (const float*)d_in[7];
    const float* W_out = (const float*)d_in[8];
    const float* b_out = (const float*)d_in[9];

    char* ws = (char*)d_ws;
    _Float16* Xh = (_Float16*)ws;                                              // 2048*128*2 = 512 KB
    _Float16* Wh = (_Float16*)(ws + (size_t)MROWS * KPAD * 2);                 // 512*128*2  = 128 KB
    float*    G1 = (float*)(ws + (size_t)MROWS * KPAD * 2 + (size_t)NGATE * KPAD * 2); // 4 MB
    float*    out = (float*)d_out;

    hipLaunchKernelGGL(k_build_x,    dim3(MROWS), dim3(256), 0, stream,
                       observed, other_paths, W_emb, b_emb, Xh);
    hipLaunchKernelGGL(k_conv_w,     dim3(256),   dim3(256), 0, stream, W_ih, Wh);
    hipLaunchKernelGGL(k_gates_gemm, dim3(512),   dim3(256), 0, stream, Xh, Wh, b_ih, G1);
    hipLaunchKernelGGL(k_scan,       dim3(1),     dim3(256), 2 * NB_BYTES, stream,
                       observed, other_paths, W_emb, b_emb, W_ih, b_ih,
                       W_hh, b_hh, W_out, b_out, G1, out);
}
